// Local2FWLUpdate_12051678233185
// MI455X (gfx1250) — matrix operand order repaired and checked by reading, not compiled, not run on hardware
//
#include <hip/hip_runtime.h>
#include <stdint.h>

// ---------------------------------------------------------------------------
// CDNA5 (gfx1250) implementation of the Local2FWL triplet update.
// All GEMMs run on v_wmma_f32_16x16x32_bf16 (wave32 WMMA), f32 accumulate.
// ---------------------------------------------------------------------------

typedef __attribute__((ext_vector_type(16))) __bf16 v16bf;
typedef __attribute__((ext_vector_type(8)))  float  v8f;

#define D        128
#define NPAIRS   100000
#define NTRIP    500000

#define TT   64     // triplets per block (triplet kernel)
#define PT   64     // pairs per block (phi kernel)
#define AK   392    // LDS row stride for 384-wide concat rows (+8 pad -> 4-bank skew)
#define HK   136    // LDS row stride for 128-wide hidden rows (+8 pad)
#define CK   264    // LDS row stride for 256-wide concat rows (+8 pad)

__device__ __forceinline__ unsigned short f2bf(float f) {
  unsigned int u = __float_as_uint(f);
  u += 0x7fffu + ((u >> 16) & 1u);       // round-to-nearest-even
  return (unsigned short)(u >> 16);
}

union Frag {
  uint4 q[2];
  v16bf v;
};

// A-fragment (16x32 bf16): lane L<16 -> row M=L, K=0..15 (contiguous);
// lane L>=16 -> row M=L-16, K=16..31.
// p points at rowbase + kstep*32 (ushort units) in LDS.
__device__ __forceinline__ v16bf load_afrag(const unsigned short* p, int lane) {
  const unsigned short* b = p + ((lane < 16) ? 0 : 16);
  Frag f;
  f.q[0] = *reinterpret_cast<const uint4*>(b);
  f.q[1] = *reinterpret_cast<const uint4*>(b + 8);
  return f.v;
}

// B-fragment (32x16 bf16) from column-major weights wt[n*Ktot + k]:
// lane L<16 -> column N, K=kstep*32+0..15 (contiguous 32B);
// lane L>=16 -> column N, K=kstep*32+16..31.
__device__ __forceinline__ v16bf load_bfrag(const unsigned short* wt, int col,
                                            int kstep, int Ktot, int lane) {
  const unsigned short* p = wt + (size_t)col * Ktot + kstep * 32 + ((lane < 16) ? 0 : 16);
  Frag f;
  f.q[0] = *reinterpret_cast<const uint4*>(p);
  f.q[1] = *reinterpret_cast<const uint4*>(p + 8);
  return f.v;
}

__device__ __forceinline__ float silu(float x) {
  return x / (1.0f + __expf(-x));
}

// ------------------------------- prep kernels ------------------------------

__global__ void convert_h_kernel(const float* __restrict__ h,
                                 unsigned short* __restrict__ out, int n) {
  int i = blockIdx.x * 256 + threadIdx.x;
  if (i < n) out[i] = f2bf(h[i]);
}

// wt[n*Kt + k] = bf16(w[k*128 + n]) for n in [0,128), k in [0,Kt)
__global__ void transpose_w_kernel(const float* __restrict__ w,
                                   unsigned short* __restrict__ wt, int Kt) {
  int i = blockIdx.x * 256 + threadIdx.x;
  if (i >= 128 * Kt) return;
  int n = i / Kt;
  int k = i - n * Kt;
  wt[i] = f2bf(w[(size_t)k * D + n]);
}

// ------------------------------ triplet kernel -----------------------------
// Per block: 64 triplets. psi layer1 (K=384 WMMA + K=4 geom via VALU), SiLU,
// psi layer2 (K=128 WMMA), atomic scatter-add into agg[vw].

__global__ __launch_bounds__(256)
void triplet_kernel(const unsigned short* __restrict__ h_bf,
                    const int* __restrict__ vu, const int* __restrict__ uw,
                    const int* __restrict__ vw,
                    const float* __restrict__ geom,
                    const unsigned short* __restrict__ w1t,   // [128][384] bf16
                    const float* __restrict__ psi_w1,         // [388][128] f32 (geom rows)
                    const float* __restrict__ b1,
                    const unsigned short* __restrict__ w2t,   // [128][128] bf16
                    const float* __restrict__ b2,
                    float* __restrict__ agg) {
  __shared__ __align__(16) unsigned short smem[TT * AK];      // 50,176 B (aliased for hidden)
  __shared__ float sgeom[TT * 4];
  __shared__ int   sidx[TT];

  unsigned short* srow = smem;       // gather buffer: [64][AK], 384 used
  unsigned short* shid = smem;       // hidden buffer (after barrier): [64][HK]

  const int tid = threadIdx.x;
  const int t0  = blockIdx.x * TT;

  // Gather 3x64 rows of 128 bf16 (16 uint4 each) -> 3072 uint4 items
  for (int it = tid; it < TT * 3 * 16; it += 256) {
    int qi  = it & 15;
    int seg = (it >> 4) % 3;
    int row = it / 48;
    int t = t0 + row; if (t >= NTRIP) t = NTRIP - 1;
    int pi = (seg == 0) ? vu[t] : (seg == 1) ? uw[t] : vw[t];
    *reinterpret_cast<uint4*>(&srow[row * AK + seg * 128 + qi * 8]) =
        reinterpret_cast<const uint4*>(h_bf)[(size_t)pi * 16 + qi];
  }
  if (tid < TT * 4) {
    int row = tid >> 2, g = tid & 3;
    int t = t0 + row;
    sgeom[tid] = (t < NTRIP) ? geom[(size_t)t * 4 + g] : 0.0f;
  }
  if (tid < TT) {
    int t = t0 + tid;
    sidx[tid] = (t < NTRIP) ? vw[t] : -1;
  }
  __syncthreads();

  const int wave    = tid >> 5;
  const int lane    = tid & 31;
  const int mt      = wave >> 1;           // M-tile 0..3
  const int nh      = wave & 1;            // N half 0..1 (4 tiles each)
  const int nlane   = lane & 15;
  const int arow    = mt * 16 + nlane;     // A-fragment source row for this lane
  const int mrow_hi = (lane >> 4) * 8;     // C/D: M = j + 8*(lane>=16)

  const v8f vz = {0.f, 0.f, 0.f, 0.f, 0.f, 0.f, 0.f, 0.f};
  v8f acc[4];
#pragma unroll
  for (int j = 0; j < 4; ++j) acc[j] = vz;

  // psi layer 1: K = 384 (12 steps of 32)
  for (int ks = 0; ks < 12; ++ks) {
    v16bf a = load_afrag(&srow[arow * AK + ks * 32], lane);
#pragma unroll
    for (int j = 0; j < 4; ++j) {
      int col = (nh * 4 + j) * 16 + nlane;
      v16bf b = load_bfrag(w1t, col, ks, 384, lane);
      acc[j] = __builtin_amdgcn_wmma_f32_16x16x32_bf16(
          false, a, false, b, (short)0, acc[j], false, false);
    }
  }
  __syncthreads();   // everyone done reading srow before hidden overwrites it

  // bias + geom contribution + SiLU -> hidden (bf16, LDS)
#pragma unroll
  for (int j = 0; j < 4; ++j) {
    int ncol = (nh * 4 + j) * 16 + nlane;
    float bias = b1[ncol];
    float gw0 = psi_w1[(size_t)384 * D + ncol];
    float gw1 = psi_w1[(size_t)385 * D + ncol];
    float gw2 = psi_w1[(size_t)386 * D + ncol];
    float gw3 = psi_w1[(size_t)387 * D + ncol];
#pragma unroll
    for (int r = 0; r < 8; ++r) {
      int row = mt * 16 + r + mrow_hi;
      float x = acc[j][r] + bias
              + gw0 * sgeom[row * 4 + 0] + gw1 * sgeom[row * 4 + 1]
              + gw2 * sgeom[row * 4 + 2] + gw3 * sgeom[row * 4 + 3];
      shid[row * HK + ncol] = f2bf(silu(x));
    }
  }
  __syncthreads();

  // psi layer 2: K = 128 (4 steps)
  v8f acc2[4];
#pragma unroll
  for (int j = 0; j < 4; ++j) acc2[j] = vz;
  for (int ks = 0; ks < 4; ++ks) {
    v16bf a = load_afrag(&shid[arow * HK + ks * 32], lane);
#pragma unroll
    for (int j = 0; j < 4; ++j) {
      int col = (nh * 4 + j) * 16 + nlane;
      v16bf b = load_bfrag(w2t, col, ks, 128, lane);
      acc2[j] = __builtin_amdgcn_wmma_f32_16x16x32_bf16(
          false, a, false, b, (short)0, acc2[j], false, false);
    }
  }

  // scatter-add m_triplet onto agg[vw]
#pragma unroll
  for (int j = 0; j < 4; ++j) {
    int ncol = (nh * 4 + j) * 16 + nlane;
    float bias = b2[ncol];
#pragma unroll
    for (int r = 0; r < 8; ++r) {
      int row  = mt * 16 + r + mrow_hi;
      int dest = sidx[row];
      if (dest >= 0)
        atomicAdd(&agg[(size_t)dest * D + ncol], acc2[j][r] + bias);
    }
  }
}

// -------------------------------- phi kernel -------------------------------
// Per block: 64 pairs. phi layer1 (K=256 WMMA on concat(h, agg)), SiLU,
// phi layer2 (K=128 WMMA), residual add, write d_out.

__global__ __launch_bounds__(256)
void phi_kernel(const float* __restrict__ h_pair,
                const unsigned short* __restrict__ h_bf,
                const float* __restrict__ agg,
                const unsigned short* __restrict__ w1t,   // [128][256] bf16
                const float* __restrict__ b1,
                const unsigned short* __restrict__ w2t,   // [128][128] bf16
                const float* __restrict__ b2,
                float* __restrict__ out) {
  __shared__ __align__(16) unsigned short sIn[PT * CK];    // 33,792 B
  __shared__ __align__(16) unsigned short shid[PT * HK];   // 17,408 B

  const int tid = threadIdx.x;
  const int p0  = blockIdx.x * PT;

  // h part: 64 rows x 16 uint4 of bf16
  for (int it = tid; it < PT * 16; it += 256) {
    int qi = it & 15, row = it >> 4;
    int p = p0 + row; if (p >= NPAIRS) p = NPAIRS - 1;
    *reinterpret_cast<uint4*>(&sIn[row * CK + qi * 8]) =
        reinterpret_cast<const uint4*>(h_bf)[(size_t)p * 16 + qi];
  }
  // agg part: 64 rows x 32 float4, convert f32 -> bf16
  for (int it = tid; it < PT * 32; it += 256) {
    int qi = it & 31, row = it >> 5;
    int p = p0 + row; if (p >= NPAIRS) p = NPAIRS - 1;
    float4 v = reinterpret_cast<const float4*>(agg)[(size_t)p * 32 + qi];
    unsigned short* dst = &sIn[row * CK + 128 + qi * 4];
    dst[0] = f2bf(v.x); dst[1] = f2bf(v.y); dst[2] = f2bf(v.z); dst[3] = f2bf(v.w);
  }
  __syncthreads();

  const int wave    = tid >> 5;
  const int lane    = tid & 31;
  const int mt      = wave >> 1;
  const int nh      = wave & 1;
  const int nlane   = lane & 15;
  const int arow    = mt * 16 + nlane;
  const int mrow_hi = (lane >> 4) * 8;

  const v8f vz = {0.f, 0.f, 0.f, 0.f, 0.f, 0.f, 0.f, 0.f};
  v8f acc[4];
#pragma unroll
  for (int j = 0; j < 4; ++j) acc[j] = vz;

  // phi layer 1: K = 256 (8 steps)
  for (int ks = 0; ks < 8; ++ks) {
    v16bf a = load_afrag(&sIn[arow * CK + ks * 32], lane);
#pragma unroll
    for (int j = 0; j < 4; ++j) {
      int col = (nh * 4 + j) * 16 + nlane;
      v16bf b = load_bfrag(w1t, col, ks, 256, lane);
      acc[j] = __builtin_amdgcn_wmma_f32_16x16x32_bf16(
          false, a, false, b, (short)0, acc[j], false, false);
    }
  }

#pragma unroll
  for (int j = 0; j < 4; ++j) {
    int ncol = (nh * 4 + j) * 16 + nlane;
    float bias = b1[ncol];
#pragma unroll
    for (int r = 0; r < 8; ++r) {
      int row = mt * 16 + r + mrow_hi;
      shid[row * HK + ncol] = f2bf(silu(acc[j][r] + bias));
    }
  }
  __syncthreads();

  // phi layer 2: K = 128 (4 steps)
  v8f acc2[4];
#pragma unroll
  for (int j = 0; j < 4; ++j) acc2[j] = vz;
  for (int ks = 0; ks < 4; ++ks) {
    v16bf a = load_afrag(&shid[arow * HK + ks * 32], lane);
#pragma unroll
    for (int j = 0; j < 4; ++j) {
      int col = (nh * 4 + j) * 16 + nlane;
      v16bf b = load_bfrag(w2t, col, ks, 128, lane);
      acc2[j] = __builtin_amdgcn_wmma_f32_16x16x32_bf16(
          false, a, false, b, (short)0, acc2[j], false, false);
    }
  }

  // residual add + store
#pragma unroll
  for (int j = 0; j < 4; ++j) {
    int ncol = (nh * 4 + j) * 16 + nlane;
    float bias = b2[ncol];
#pragma unroll
    for (int r = 0; r < 8; ++r) {
      int row = mt * 16 + r + mrow_hi;
      int p = p0 + row;
      if (p < NPAIRS) {
        size_t o = (size_t)p * D + ncol;
        out[o] = h_pair[o] + acc2[j][r] + bias;
      }
    }
  }
}

// ------------------------------- host launch -------------------------------

extern "C" void kernel_launch(void* const* d_in, const int* in_sizes, int n_in,
                              void* d_out, int out_size, void* d_ws, size_t ws_size,
                              hipStream_t stream) {
  const float* h_pair  = (const float*)d_in[0];
  const int*   vu      = (const int*)d_in[1];
  const int*   uw      = (const int*)d_in[2];
  const int*   vw      = (const int*)d_in[3];
  const float* geom    = (const float*)d_in[4];
  const float* psi_w1  = (const float*)d_in[5];
  const float* psi_b1  = (const float*)d_in[6];
  const float* psi_w2  = (const float*)d_in[7];
  const float* psi_b2  = (const float*)d_in[8];
  const float* phi_w1  = (const float*)d_in[9];
  const float* phi_b1  = (const float*)d_in[10];
  const float* phi_w2  = (const float*)d_in[11];
  const float* phi_b2  = (const float*)d_in[12];
  float* out = (float*)d_out;

  char* ws = (char*)d_ws;
  size_t off = 0;
  auto alloc = [&](size_t bytes) -> void* {
    void* p = ws + off;
    off = (off + bytes + 255) & ~(size_t)255;
    return p;
  };
  float*          agg      = (float*)alloc((size_t)NPAIRS * D * sizeof(float));
  unsigned short* h_bf     = (unsigned short*)alloc((size_t)NPAIRS * D * 2);
  unsigned short* psi_w1t  = (unsigned short*)alloc((size_t)128 * 384 * 2);
  unsigned short* psi_w2t  = (unsigned short*)alloc((size_t)128 * 128 * 2);
  unsigned short* phi_w1t  = (unsigned short*)alloc((size_t)128 * 256 * 2);
  unsigned short* phi_w2t  = (unsigned short*)alloc((size_t)128 * 128 * 2);

  hipMemsetAsync(agg, 0, (size_t)NPAIRS * D * sizeof(float), stream);

  const int nh = NPAIRS * D;
  convert_h_kernel<<<(nh + 255) / 256, 256, 0, stream>>>(h_pair, h_bf, nh);
  transpose_w_kernel<<<(128 * 384 + 255) / 256, 256, 0, stream>>>(psi_w1, psi_w1t, 384);
  transpose_w_kernel<<<(128 * 128 + 255) / 256, 256, 0, stream>>>(psi_w2, psi_w2t, 128);
  transpose_w_kernel<<<(128 * 256 + 255) / 256, 256, 0, stream>>>(phi_w1, phi_w1t, 256);
  transpose_w_kernel<<<(128 * 128 + 255) / 256, 256, 0, stream>>>(phi_w2, phi_w2t, 128);

  triplet_kernel<<<(NTRIP + TT - 1) / TT, 256, 0, stream>>>(
      h_bf, vu, uw, vw, geom, psi_w1t, psi_w1, psi_b1, psi_w2t, psi_b2, agg);

  phi_kernel<<<(NPAIRS + PT - 1) / PT, 256, 0, stream>>>(
      h_pair, h_bf, agg, phi_w1t, phi_b1, phi_w2t, phi_b2, out);

  (void)in_sizes; (void)n_in; (void)out_size; (void)ws_size;
}
